// LegislativeGraphEncoder_44994077392974
// MI455X (gfx1250) — matrix operand order repaired and checked by reading, not compiled, not run on hardware
//
#include <hip/hip_runtime.h>
#include <math.h>

// ---------------------------------------------------------------- types ----
typedef __attribute__((ext_vector_type(16))) __bf16          v16bf;
typedef __attribute__((ext_vector_type(16))) unsigned short  v16u16;
typedef __attribute__((ext_vector_type(8)))  unsigned int    v8u;
typedef __attribute__((ext_vector_type(8)))  float           v8f;

#define D_MODEL 192
#define NHEAD   4
#define DK      48

__device__ __forceinline__ unsigned int f2bf_u(float x) {
    // round-to-nearest-even f32 -> bf16 (as low 16 bits of uint)
    unsigned int u = __float_as_uint(x);
    unsigned int r = u + 0x7FFFu + ((u >> 16) & 1u);
    return r >> 16;
}

__device__ __forceinline__ float gelu_exact(float x) {
    return 0.5f * x * (1.0f + erff(x * 0.70710678118654752f));
}

// ---------------------------------------------------- WMMA GEMM (bf16) ----
// C[M x N] = epi( A[M x K](f32,lda) * B[K x N](f32,ldb) )
// Block tile 128x64, K-step 32, 4 waves; each wave owns a 32-row strip and
// issues 8x v_wmma_f32_16x16x32_bf16 per K step (B fragments reused).
// LDS tiles are stored in *fragment order*: [frag][lane][8 dwords], so a
// fragment load is one contiguous 32B read per lane (2x ds_load_b128).
// Ping-pong LDS buffers overlap global staging with WMMA.
// act: 0 none, 1 GELU, 2 ReLU.  Epilogue order: +bias -> act -> +resid.
#define BM 128
#define BN 64
#define BK 32

__global__ __launch_bounds__(128)
void gemm_wmma_kernel(const float* __restrict__ A, int lda,
                      const float* __restrict__ B, int ldb,
                      float* __restrict__ C, int ldc,
                      int M, int N, int K,
                      const float* __restrict__ bias,
                      const float* __restrict__ resid,
                      int act)
{
    // [buf][strip 0..7][lane 0..31][dword 0..7]  : A fragments (16x32 bf16)
    __shared__ alignas(32) unsigned int AfragU[2][8][32][8];
    // [buf][tile 0..3][lane 0..31][dword 0..7]   : B fragments (32x16 bf16)
    __shared__ alignas(32) unsigned int BfragU[2][4][32][8];

    const int tid  = threadIdx.x;
    const int wave = tid >> 5;
    const int lane = tid & 31;
    const int rowBlock = blockIdx.x * BM;
    const int colBlock = blockIdx.y * BN;

    // ---- staging helpers (global f32 -> packed bf16x2 regs -> LDS dwords)
    unsigned int Areg[16], Breg[8];

    auto stageA = [&](int k0) {
#pragma unroll
        for (int j = 0; j < 16; ++j) {
            int p  = j * 128 + tid;           // pair index in 128x32 tile
            int r  = p >> 4;                  // row in tile
            int c  = (p & 15) * 2;            // even k within tile
            int gr = rowBlock + r, gc = k0 + c;
            float v0 = (gr < M && gc     < K) ? A[(size_t)gr * lda + gc]     : 0.0f;
            float v1 = (gr < M && gc + 1 < K) ? A[(size_t)gr * lda + gc + 1] : 0.0f;
            Areg[j] = f2bf_u(v0) | (f2bf_u(v1) << 16);
        }
    };
    auto storeA = [&](int buf) {
        unsigned int* dst = &AfragU[buf][0][0][0];
#pragma unroll
        for (int j = 0; j < 16; ++j) {
            int p = j * 128 + tid;
            int r = p >> 4;
            int c = (p & 15) * 2;
            int strip = r >> 4, m = r & 15;
            int ln = m + (((c >> 3) & 1) << 4);          // fragment lane
            int jj = (c & 7) + ((c & 16) ? 8 : 0);       // element index
            dst[(strip * 32 + ln) * 8 + (jj >> 1)] = Areg[j];
        }
    };
    auto stageB = [&](int k0) {
#pragma unroll
        for (int j = 0; j < 8; ++j) {
            int p = j * 128 + tid;            // pair index in 32x64 tile
            int n = p & 63;
            int k = (p >> 6) * 2;             // even k within tile
            int gk = k0 + k;
            const float* bp = B + (size_t)gk * ldb + colBlock + n;
            float v0 = (gk     < K) ? bp[0]   : 0.0f;
            float v1 = (gk + 1 < K) ? bp[ldb] : 0.0f;
            Breg[j] = f2bf_u(v0) | (f2bf_u(v1) << 16);
        }
    };
    auto storeB = [&](int buf) {
        unsigned int* dst = &BfragU[buf][0][0][0];
#pragma unroll
        for (int j = 0; j < 8; ++j) {
            int p = j * 128 + tid;
            int n = p & 63;
            int k = (p >> 6) * 2;
            int tile = n >> 4;
            int ln = (n & 15) + ((k & 8) ? 16 : 0);
            dst[(tile * 32 + ln) * 8 + (((k & 7) + ((k & 16) ? 8 : 0)) >> 1)] = Breg[j];
        }
    };

    v8f acc[2][4];
    const v8f vzero = {0.f, 0.f, 0.f, 0.f, 0.f, 0.f, 0.f, 0.f};
#pragma unroll
    for (int h = 0; h < 2; ++h)
#pragma unroll
        for (int t = 0; t < 4; ++t) acc[h][t] = vzero;

    // ---- pipelined main loop
    stageA(0); stageB(0);
    storeA(0); storeB(0);
    int buf = 0;
    for (int k0 = 0; k0 < K; k0 += BK) {
        __syncthreads();                      // LDS[buf] ready
        const bool more = (k0 + BK) < K;
        if (more) { stageA(k0 + BK); stageB(k0 + BK); }

        v16bf a0 = __builtin_bit_cast(v16bf,
                     *(const v8u*)&AfragU[buf][2 * wave][lane][0]);
        v16bf a1 = __builtin_bit_cast(v16bf,
                     *(const v8u*)&AfragU[buf][2 * wave + 1][lane][0]);
#pragma unroll
        for (int t = 0; t < 4; ++t) {
            v16bf bf = __builtin_bit_cast(v16bf,
                         *(const v8u*)&BfragU[buf][t][lane][0]);
            acc[0][t] = __builtin_amdgcn_wmma_f32_16x16x32_bf16(
                false, a0, false, bf, (short)0, acc[0][t], false, false);
            acc[1][t] = __builtin_amdgcn_wmma_f32_16x16x32_bf16(
                false, a1, false, bf, (short)0, acc[1][t], false, false);
        }

        if (more) {
            __syncthreads();                  // everyone done reading buf^1 era
            storeA(buf ^ 1); storeB(buf ^ 1);
            buf ^= 1;
        }
    }

    // ---- store: C 16x16 f32 layout: VGPR r -> M=r(+8 for lanes>=16), N=lane&15
    const int n_in  = lane & 15;
    const int mhalf = (lane >> 4) * 8;
#pragma unroll
    for (int h = 0; h < 2; ++h) {
#pragma unroll
        for (int t = 0; t < 4; ++t) {
            int col = colBlock + t * 16 + n_in;
#pragma unroll
            for (int r = 0; r < 8; ++r) {
                int row = rowBlock + wave * 32 + h * 16 + mhalf + r;
                if (row < M) {
                    float v = acc[h][t][r];
                    if (bias)  v += bias[col];
                    if (act == 1)      v = gelu_exact(v);
                    else if (act == 2) v = fmaxf(v, 0.0f);
                    if (resid) v += resid[(size_t)row * ldc + col];
                    C[(size_t)row * ldc + col] = v;
                }
            }
        }
    }
}

// -------------------------------------------- LayerNorm (+t2v, +relu) ----
// One wave per row. If ts != null, columns [Kx, K) are Time2Vec(ts[row]).
__global__ __launch_bounds__(256)
void ln_rows_kernel(const float* __restrict__ x, int Kx, int ldx,
                    const float* __restrict__ ts,
                    const float* __restrict__ t2v_w0,
                    const float* __restrict__ t2v_w,
                    const float* __restrict__ t2v_b,
                    const float* __restrict__ g, const float* __restrict__ b,
                    float* __restrict__ out, int ldo,
                    int N, int K, int relu_out)
{
    const int wave = threadIdx.x >> 5;
    const int lane = threadIdx.x & 31;
    const int row  = blockIdx.x * 8 + wave;
    if (row >= N) return;
    const float t = ts ? ts[row] : 0.0f;

    float s = 0.0f, ss = 0.0f;
    for (int c = lane; c < K; c += 32) {
        float v;
        if (c < Kx) v = x[(size_t)row * ldx + c];
        else {
            int j = c - Kx;
            v = (j == 0) ? (t2v_w0[0] * t + t2v_b[0])
                         : sinf(t2v_w[j - 1] * t + t2v_b[j]);
        }
        s += v; ss += v * v;
    }
#pragma unroll
    for (int off = 16; off > 0; off >>= 1) {
        s  += __shfl_xor(s,  off);
        ss += __shfl_xor(ss, off);
    }
    const float mean = s / (float)K;
    const float var  = ss / (float)K - mean * mean;
    const float rstd = rsqrtf(var + 1e-5f);

    for (int c = lane; c < K; c += 32) {
        float v;
        if (c < Kx) v = x[(size_t)row * ldx + c];
        else {
            int j = c - Kx;
            v = (j == 0) ? (t2v_w0[0] * t + t2v_b[0])
                         : sinf(t2v_w[j - 1] * t + t2v_b[j]);
        }
        float o = (v - mean) * rstd * g[c] + b[c];
        if (relu_out) o = fmaxf(o, 0.0f);
        out[(size_t)row * ldo + c] = o;
    }
}

// ------------------------------------------------- elementwise add ----
__global__ __launch_bounds__(256)
void add_kernel(float* __restrict__ a, const float* __restrict__ b, size_t n)
{
    size_t i = (size_t)blockIdx.x * blockDim.x + threadIdx.x;
    if (i < n) a[i] += b[i];
}

// ------------------------------------------- per-edge head attention ----
// One wave per edge; lane l owns channels [6l, 6l+6) -> head = l/8.
__global__ __launch_bounds__(256)
void edge_attn_kernel(const float* __restrict__ qbuf,
                      const float* __restrict__ kbuf,
                      const float* __restrict__ vbuf,
                      const int* __restrict__ src, const int* __restrict__ dst,
                      const float* __restrict__ ew,
                      const float* __restrict__ relp,   // [H*DK]=192 floats
                      float* __restrict__ msg, int E)
{
    const int wave = threadIdx.x >> 5;
    const int lane = threadIdx.x & 31;
    const int e = blockIdx.x * 8 + wave;
    if (e >= E) return;

    const int s = src[e], d = dst[e];
    const float* q = qbuf + (size_t)d * D_MODEL;
    const float* k = kbuf + (size_t)s * D_MODEL;
    const float* v = vbuf + (size_t)s * D_MODEL;

    const int base = lane * 6;
    const int h    = lane >> 3;
    float part = 0.0f;
    float vv[6];
#pragma unroll
    for (int j = 0; j < 6; ++j) {
        int dd = base + j;
        part += q[dd] * (k[dd] + relp[dd]);
        vv[j] = v[dd];
    }
    part += __shfl_xor(part, 1);
    part += __shfl_xor(part, 2);
    part += __shfl_xor(part, 4);

    const float logit = part * 0.14433756729740643f /* 1/sqrt(48) */ + ew[e];
    const float l0 = __shfl(logit, 0),  l1 = __shfl(logit, 8);
    const float l2 = __shfl(logit, 16), l3 = __shfl(logit, 24);
    const float mx = fmaxf(fmaxf(l0, l1), fmaxf(l2, l3));
    const float e0 = __expf(l0 - mx), e1 = __expf(l1 - mx);
    const float e2 = __expf(l2 - mx), e3 = __expf(l3 - mx);
    const float denom = e0 + e1 + e2 + e3;
    const float p = ((h == 0) ? e0 : (h == 1) ? e1 : (h == 2) ? e2 : e3) / denom;

    float* mrow = msg + (size_t)d * D_MODEL;
#pragma unroll
    for (int j = 0; j < 6; ++j)
        unsafeAtomicAdd(&mrow[base + j], p * vv[j]);
}

// --------------------------------------------- vote-conv edge scatter ----
__global__ __launch_bounds__(256)
void edge_vote_kernel(const float* __restrict__ hlt,
                      const float* __restrict__ ef,     // chunk-local rows
                      const float* __restrict__ vote_attr,
                      const int* __restrict__ src, const int* __restrict__ dst,
                      float* __restrict__ hbv, int e0, int e1)
{
    const int wave = threadIdx.x >> 5;
    const int lane = threadIdx.x & 31;
    const int e = e0 + blockIdx.x * 8 + wave;
    if (e >= e1) return;

    float pol = vote_attr[(size_t)e * 385];
    pol = fminf(fmaxf(pol, 0.0f), 1.0f) + 0.01f;

    const int s = src[e], d = dst[e];
    const float* hs  = hlt + (size_t)s * D_MODEL;
    const float* efr = ef  + (size_t)(e - e0) * D_MODEL;
    float* outp = hbv + (size_t)d * D_MODEL;

    const int base = lane * 6;
#pragma unroll
    for (int j = 0; j < 6; ++j)
        unsafeAtomicAdd(&outp[base + j], hs[base + j] * efr[base + j] * pol);
}

// ---------------------------------------------------------------- host ----
static inline void launch_gemm(const float* A, int lda, const float* B, int ldb,
                               float* C, int ldc, int M, int N, int K,
                               const float* bias, const float* resid, int act,
                               hipStream_t s)
{
    dim3 grid((M + BM - 1) / BM, N / BN), block(128);
    gemm_wmma_kernel<<<grid, block, 0, s>>>(A, lda, B, ldb, C, ldc,
                                            M, N, K, bias, resid, act);
}

extern "C" void kernel_launch(void* const* d_in, const int* in_sizes, int n_in,
                              void* d_out, int out_size, void* d_ws, size_t ws_size,
                              hipStream_t stream)
{
    (void)in_sizes; (void)n_in; (void)out_size; (void)ws_size;

    // ---- node-type metadata (ORDER = lt,bill,bv,comm,lob,don,top)
    static const int Ncnt[7]   = {20000, 50000, 100000, 2000, 10000, 50000, 5000};
    static const int INDv[7]   = {769, 389, 385, 2, 385, 384, 384};
    static const int DINv[7]   = {781, 401, 397, 2, 385, 384, 384};
    static const int rowOff[7] = {0, 20000, 70000, 170000, 172000, 182000, 232000};
    const int NTOT = 237000;

    // ---- inputs
    const float* x[7]; for (int i = 0; i < 7; ++i) x[i] = (const float*)d_in[i];
    const float* tsv[7] = {(const float*)d_in[7], (const float*)d_in[8],
                           (const float*)d_in[9], nullptr, nullptr, nullptr, nullptr};
    const float* t2v_w0 = (const float*)d_in[10];
    const float* t2v_w  = (const float*)d_in[11];
    const float* t2v_b  = (const float*)d_in[12];
    const float* Qw  = (const float*)d_in[34];
    const float* Kw  = (const float*)d_in[35];
    const float* Vw  = (const float*)d_in[36];
    const float* rel = (const float*)d_in[37];
    const float* ffn_g  = (const float*)d_in[38];
    const float* ffn_b  = (const float*)d_in[39];
    const float* ffn_w1 = (const float*)d_in[40];
    const float* ffn_w2 = (const float*)d_in[41];
    const float* vote_attr = (const float*)d_in[46];
    const float* vmlp_w1 = (const float*)d_in[47];
    const float* vmlp_b1 = (const float*)d_in[48];
    const float* vmlp_w2 = (const float*)d_in[49];
    const float* vmlp_b2 = (const float*)d_in[50];
    const float* norm_g  = (const float*)d_in[51];
    const float* norm_b  = (const float*)d_in[52];

    // ---- workspace layout
    const size_t SH = (size_t)NTOT * D_MODEL;             // elements
    char* base = (char*)d_ws;
    float* H   = (float*)base;
    float* MSG = (float*)(base + SH * sizeof(float));
    float* S1  = (float*)(base + 2 * SH * sizeof(float));               // 256 MB
    float* S2  = (float*)(base + 2 * SH * sizeof(float) + (size_t)268435456); // 64 MB

    // ========================== 1) feature projection ==========================
    for (int i = 0; i < 7; ++i) {
        const int N = Ncnt[i], Kx = INDv[i], K = DINv[i];
        const float* pg = (const float*)d_in[13 + 3 * i];
        const float* pb = (const float*)d_in[14 + 3 * i];
        const float* pw = (const float*)d_in[15 + 3 * i];
        dim3 lg((N + 7) / 8), lb(256);
        ln_rows_kernel<<<lg, lb, 0, stream>>>(x[i], Kx, Kx, tsv[i],
                                              t2v_w0, t2v_w, t2v_b,
                                              pg, pb, S1, K, N, K, 0);
        launch_gemm(S1, K, pw, D_MODEL, H + (size_t)rowOff[i] * D_MODEL, D_MODEL,
                    N, D_MODEL, K, nullptr, nullptr, /*GELU*/1, stream);
    }

    // ---- relations: {stOff, dtOff, Nst, Ndt, ri, srcIdx, dstIdx, ewIdx, E}
    struct RelT { int stOff, dtOff, Nst, Ndt, ri, si, di, ei, E; };
    static const RelT rels[4] = {
        {0,      70000, 20000, 100000, 0, 53, 54, 42, 300000},   // lt   -> bv
        {20000,  70000, 50000, 100000, 1, 55, 56, 43, 150000},   // bill -> bv
        {170000, 20000,  2000,  50000, 2, 57, 58, 44, 300000},   // comm -> bill
        {182000, 0,     50000,  20000, 3, 59, 60, 45, 400000},   // don  -> lt
    };
    const size_t QCAP = (size_t)100000 * D_MODEL;  // qbuf capacity (elements)
    const size_t KCAP = (size_t)50000  * D_MODEL;

    // ========================== 2) NL = 3 transformer layers ====================
    for (int L = 0; L < 3; ++L) {
        hipMemsetAsync(MSG, 0, SH * sizeof(float), stream);

        for (int r = 0; r < 4; ++r) {
            const RelT& R = rels[r];
            float* qbuf = S1;
            float* kbuf = S1 + QCAP;
            float* vbuf = S1 + QCAP + KCAP;
            const float* qwL = Qw + (size_t)L * D_MODEL * D_MODEL;
            const float* kwL = Kw + (size_t)L * D_MODEL * D_MODEL;
            const float* vwL = Vw + (size_t)L * D_MODEL * D_MODEL;

            launch_gemm(H + (size_t)R.dtOff * D_MODEL, D_MODEL, qwL, D_MODEL,
                        qbuf, D_MODEL, R.Ndt, D_MODEL, D_MODEL,
                        nullptr, nullptr, 0, stream);
            launch_gemm(H + (size_t)R.stOff * D_MODEL, D_MODEL, kwL, D_MODEL,
                        kbuf, D_MODEL, R.Nst, D_MODEL, D_MODEL,
                        nullptr, nullptr, 0, stream);
            launch_gemm(H + (size_t)R.stOff * D_MODEL, D_MODEL, vwL, D_MODEL,
                        vbuf, D_MODEL, R.Nst, D_MODEL, D_MODEL,
                        nullptr, nullptr, 0, stream);

            const float* relp = rel + ((size_t)L * 4 + R.ri) * (NHEAD * DK);
            dim3 eg((R.E + 7) / 8), eb(256);
            edge_attn_kernel<<<eg, eb, 0, stream>>>(
                qbuf, kbuf, vbuf,
                (const int*)d_in[R.si], (const int*)d_in[R.di],
                (const float*)d_in[R.ei], relp,
                MSG + (size_t)R.dtOff * D_MODEL, R.E);
        }

        // h += msg (residual), then FFN shared across all node types
        {
            size_t n = SH;
            dim3 ag((unsigned)((n + 255) / 256)), ab(256);
            add_kernel<<<ag, ab, 0, stream>>>(H, MSG, n);
        }
        const int CH = 65536;
        for (int r0 = 0; r0 < NTOT; r0 += CH) {
            const int cn = (NTOT - r0 < CH) ? (NTOT - r0) : CH;
            dim3 lg((cn + 7) / 8), lb(256);
            ln_rows_kernel<<<lg, lb, 0, stream>>>(
                H + (size_t)r0 * D_MODEL, D_MODEL, D_MODEL, nullptr,
                t2v_w0, t2v_w, t2v_b,
                ffn_g + (size_t)L * D_MODEL, ffn_b + (size_t)L * D_MODEL,
                S2, D_MODEL, cn, D_MODEL, 0);
            launch_gemm(S2, D_MODEL, ffn_w1 + (size_t)L * D_MODEL * 768, 768,
                        S1, 768, cn, 768, D_MODEL, nullptr, nullptr, /*GELU*/1, stream);
            launch_gemm(S1, 768, ffn_w2 + (size_t)L * 768 * D_MODEL, D_MODEL,
                        H + (size_t)r0 * D_MODEL, D_MODEL, cn, D_MODEL, 768,
                        nullptr, H + (size_t)r0 * D_MODEL, 0, stream);
        }
    }

    // ========================== 3) polarity-aware vote conv =====================
    {
        const int EV = 300000, CE = 75000;
        const int* srcv = (const int*)d_in[53];
        const int* dstv = (const int*)d_in[54];
        for (int e0 = 0; e0 < EV; e0 += CE) {
            const int cn = (EV - e0 < CE) ? (EV - e0) : CE;
            // ef1 = relu(attr[:,1:385] @ W1 + b1)
            launch_gemm(vote_attr + (size_t)e0 * 385 + 1, 385, vmlp_w1, D_MODEL,
                        S2, D_MODEL, cn, D_MODEL, 384, vmlp_b1, nullptr, /*RELU*/2, stream);
            // ef  = ef1 @ W2 + b2
            launch_gemm(S2, D_MODEL, vmlp_w2, D_MODEL,
                        S1, D_MODEL, cn, D_MODEL, D_MODEL, vmlp_b2, nullptr, 0, stream);
            dim3 eg((cn + 7) / 8), eb(256);
            edge_vote_kernel<<<eg, eb, 0, stream>>>(
                H /* lt rows start at 0 */, S1, vote_attr, srcv, dstv,
                H + (size_t)70000 * D_MODEL /* bv rows */, e0, e0 + cn);
        }
    }

    // ========================== 4) final per-type LN + ReLU =====================
    for (int i = 0; i < 7; ++i) {
        const int N = Ncnt[i];
        dim3 lg((N + 7) / 8), lb(256);
        ln_rows_kernel<<<lg, lb, 0, stream>>>(
            H + (size_t)rowOff[i] * D_MODEL, D_MODEL, D_MODEL, nullptr,
            t2v_w0, t2v_w, t2v_b,
            norm_g + (size_t)i * D_MODEL, norm_b + (size_t)i * D_MODEL,
            (float*)d_out + (size_t)rowOff[i] * D_MODEL, D_MODEL,
            N, D_MODEL, /*relu*/1);
    }
}
